// GraphAttentionLayer_71588514890081
// MI455X (gfx1250) — matrix operand order repaired and checked by reading, not compiled, not run on hardware
//
#include <hip/hip_runtime.h>
#include <hip/hip_bf16.h>

// ---------------------------------------------------------------------------
// GAT layer, fused flash-attention style for gfx1250 (MI455X).
//   N = 8192, F_in = 128, F_out = 64
//   out = elu( softmax_row( mask(adj, lrelu(f1_i + f2_j)) ) @ h )
// Bottleneck: streaming 256MB adj once (~11us at 23.3 TB/s). adj is loaded
// non-temporal so the reused operands (hT: 1MB f16, f2: 32KB) stay resident
// in the 192MB L2. Attention probabilities are generated in-register in WMMA
// A-fragment layout and multiplied against a pre-transposed f16 copy of h
// with v_wmma_f32_16x16x32_f16 (f32 accumulate).
// ---------------------------------------------------------------------------

typedef __attribute__((ext_vector_type(16))) _Float16 v16h;
typedef __attribute__((ext_vector_type(8)))  float    v8f;
typedef __attribute__((ext_vector_type(4)))  int      v4i;   // native clang vec
typedef __attribute__((ext_vector_type(4)))  float    v4f;   // native clang vec

#define GAT_N     8192
#define GAT_FIN   128
#define GAT_FOUT  64
#define GAT_ALPHA 0.2f
#define TILE_K    32
#define SPLITS    8
#define ROWS      16

// ---------------------------------------------------------------------------
// Kernel 1: h = x @ W ; f1 = h.a1 ; f2 = h.a2 ; store hT (f16, column-major)
// grid = N blocks of 64 threads (one block per row, one thread per out col)
// ---------------------------------------------------------------------------
__global__ __launch_bounds__(64) void gat_proj_kernel(
    const float* __restrict__ x, const float* __restrict__ W,
    const float* __restrict__ a, _Float16* __restrict__ hT,
    float* __restrict__ f1, float* __restrict__ f2)
{
    __shared__ float xs[GAT_FIN];
    __shared__ float r1[GAT_FOUT];
    __shared__ float r2[GAT_FOUT];

    const int row = blockIdx.x;
    const int t   = threadIdx.x;          // output column 0..63

    xs[t]      = x[(size_t)row * GAT_FIN + t];
    xs[t + 64] = x[(size_t)row * GAT_FIN + 64 + t];
    __syncthreads();

    float acc = 0.f;
#pragma unroll 8
    for (int k = 0; k < GAT_FIN; ++k)
        acc = fmaf(xs[k], W[k * GAT_FOUT + t], acc);

    hT[(size_t)t * GAT_N + row] = (_Float16)acc;

    r1[t] = acc * a[t];
    r2[t] = acc * a[GAT_FOUT + t];
    __syncthreads();
#pragma unroll
    for (int s = 32; s > 0; s >>= 1) {
        if (t < s) { r1[t] += r1[t + s]; r2[t] += r2[t + s]; }
        __syncthreads();
    }
    if (t == 0) { f1[row] = r1[0]; f2[row] = r2[0]; }
}

// ---------------------------------------------------------------------------
// Kernel 2: gmax = max_j f2[j]   (single block)
// ---------------------------------------------------------------------------
__global__ __launch_bounds__(256) void gat_gmax_kernel(
    const float* __restrict__ f2, float* __restrict__ gmax)
{
    __shared__ float red[256];
    const int t = threadIdx.x;
    float m = -3.0e38f;
    for (int i = t; i < GAT_N; i += 256) m = fmaxf(m, f2[i]);
    red[t] = m;
    __syncthreads();
#pragma unroll
    for (int s = 128; s > 0; s >>= 1) {
        if (t < s) red[t] = fmaxf(red[t], red[t + s]);
        __syncthreads();
    }
    if (t == 0) *gmax = red[0];
}

// ---------------------------------------------------------------------------
// Kernel 3: fused masked-softmax attention + (att @ h) + elu.
// One block = 256 threads = 8 waves; block owns a 16-row tile; wave w owns
// columns [w*1024, (w+1)*1024). Shift s_i = lrelu(f1_i + gmax) is the exact
// unmasked row max (lrelu monotone) -> exponents <= 0, partials add directly.
// ---------------------------------------------------------------------------
__device__ __forceinline__ float gat_p(int aij, float f2v, float f1m, float sh)
{
    float e = f1m + f2v;
    e = e > 0.f ? e : GAT_ALPHA * e;          // leaky_relu
    return aij > 0 ? __expf(e - sh) : 0.f;    // masked exp(e - rowmax)
}

__global__ __launch_bounds__(256) void gat_attn_kernel(
    const int*      __restrict__ adj,
    const _Float16* __restrict__ hT,
    const float*    __restrict__ f1,
    const float*    __restrict__ f2,
    const float*    __restrict__ gmaxp,
    float*          __restrict__ out)
{
    __shared__ float acc_lds[SPLITS][ROWS][GAT_FOUT];  // 32 KB
    __shared__ float lpart[SPLITS][32];                // per-lane denom partials
    __shared__ float l_lds[ROWS];                      // final row denominators

    const int t      = threadIdx.x;
    const int wave   = t >> 5;          // j-split id
    const int lane   = t & 31;
    const int hi     = lane >> 4;       // lane half (WMMA K-grouping)
    const int mloc   = lane & 15;       // row within tile (A-frag layout)
    const int r0     = blockIdx.x * ROWS;
    const int row    = r0 + mloc;

    const float gmax = *gmaxp;
    const float f1m  = f1[row];
    float sh = f1m + gmax;
    sh = sh > 0.f ? sh : GAT_ALPHA * sh;   // analytic per-row softmax shift

    v8f cacc[4];
#pragma unroll
    for (int g = 0; g < 4; ++g) cacc[g] = (v8f){};
    float lsum = 0.f;

    const size_t arow = (size_t)row * GAT_N;
    const int    kofs = hi * 8;               // A-frag K base for this lane half
    const int    jbeg = wave * (GAT_N / SPLITS);
    const int    jend = jbeg + (GAT_N / SPLITS);

    for (int j = jbeg; j < jend; j += TILE_K) {
        // ---- build A fragment (16x32 f16, documented wave32 layout) ----
        // adj is a 256MB one-shot stream: load non-temporal so hT/f2 stay in L2.
        union { v16h v; _Float16 e[16]; } af;
#pragma unroll
        for (int run = 0; run < 2; ++run) {
            const int base = j + kofs + run * 16;  // K runs: kofs..+7, kofs+16..+23
            const v4i A0 = __builtin_nontemporal_load((const v4i*)(adj + arow + base));
            const v4i A1 = __builtin_nontemporal_load((const v4i*)(adj + arow + base + 4));
            const v4f F0 = *(const v4f*)(f2 + base);
            const v4f F1 = *(const v4f*)(f2 + base + 4);

            const float p0 = gat_p(A0.x, F0.x, f1m, sh);
            const float p1 = gat_p(A0.y, F0.y, f1m, sh);
            const float p2 = gat_p(A0.z, F0.z, f1m, sh);
            const float p3 = gat_p(A0.w, F0.w, f1m, sh);
            const float p4 = gat_p(A1.x, F1.x, f1m, sh);
            const float p5 = gat_p(A1.y, F1.y, f1m, sh);
            const float p6 = gat_p(A1.z, F1.z, f1m, sh);
            const float p7 = gat_p(A1.w, F1.w, f1m, sh);
            lsum += ((p0 + p1) + (p2 + p3)) + ((p4 + p5) + (p6 + p7));

            const int o = run * 8;
            af.e[o + 0] = (_Float16)p0; af.e[o + 1] = (_Float16)p1;
            af.e[o + 2] = (_Float16)p2; af.e[o + 3] = (_Float16)p3;
            af.e[o + 4] = (_Float16)p4; af.e[o + 5] = (_Float16)p5;
            af.e[o + 6] = (_Float16)p6; af.e[o + 7] = (_Float16)p7;
        }

        // ---- B fragments from transposed h; 2x b128 per lane per group ----
#pragma unroll
        for (int g = 0; g < 4; ++g) {
            union { v16h v; uint4 q[2]; } bf;
            const _Float16* bp =
                hT + (size_t)(g * 16 + mloc) * GAT_N + j + hi * 8;
            bf.q[0] = *(const uint4*)(bp);
            bf.q[1] = *(const uint4*)(bp + 16);
            cacc[g] = __builtin_amdgcn_wmma_f32_16x16x32_f16(
                false, af.v, false, bf.v, (short)0, cacc[g], false, false);
        }
    }

    // ---- deterministic cross-wave combine through LDS ----
    lpart[wave][lane] = lsum;
#pragma unroll
    for (int g = 0; g < 4; ++g)
#pragma unroll
        for (int v = 0; v < 8; ++v)
            acc_lds[wave][v + 8 * hi][g * 16 + mloc] = cacc[g][v];
    __syncthreads();

    if (t < ROWS) {
        float L = 0.f;
#pragma unroll
        for (int s = 0; s < SPLITS; ++s)
            L += lpart[s][t] + lpart[s][t + 16];
        l_lds[t] = L;
    }
    __syncthreads();

    // ---- normalize, elu, coalesced non-temporal store (4 elems/thread) ----
#pragma unroll
    for (int i = 0; i < (ROWS * GAT_FOUT) / 256; ++i) {
        const int idx = t + 256 * i;
        const int rr  = idx >> 6;
        const int cc  = idx & 63;
        float v = 0.f;
#pragma unroll
        for (int s = 0; s < SPLITS; ++s) v += acc_lds[s][rr][cc];
        v /= l_lds[rr];
        v = v > 0.f ? v : (__expf(v) - 1.f);   // elu
        __builtin_nontemporal_store(v, out + (size_t)(r0 + rr) * GAT_FOUT + cc);
    }
}

// ---------------------------------------------------------------------------
// Host launch
// ---------------------------------------------------------------------------
extern "C" void kernel_launch(void* const* d_in, const int* in_sizes, int n_in,
                              void* d_out, int out_size, void* d_ws, size_t ws_size,
                              hipStream_t stream)
{
    const float* x   = (const float*)d_in[0];   // [N, F_in]
    const int*   adj = (const int*)  d_in[1];   // [N, N]
    const float* W   = (const float*)d_in[2];   // [F_in, F_out]
    const float* a   = (const float*)d_in[3];   // [2*F_out, 1]
    float*       out = (float*)d_out;           // [N, F_out]

    char* ws = (char*)d_ws;
    _Float16* hT   = (_Float16*)ws;                                   // 1 MB
    float*    f1   = (float*)(ws + (size_t)GAT_FOUT * GAT_N * 2);     // 32 KB
    float*    f2   = f1 + GAT_N;                                      // 32 KB
    float*    gmax = f2 + GAT_N;                                      // 4 B

    gat_proj_kernel<<<GAT_N, 64, 0, stream>>>(x, W, a, hT, f1, f2);
    gat_gmax_kernel<<<1, 256, 0, stream>>>(f2, gmax);
    gat_attn_kernel<<<GAT_N / ROWS, 256, 0, stream>>>(adj, hT, f1, f2, gmax, out);
}
